// QLinear_68470368633330
// MI455X (gfx1250) — matrix operand order repaired and checked by reading, not compiled, not run on hardware
//
#include <hip/hip_runtime.h>
#include <hip/hip_bf16.h>

// ---------------------------------------------------------------------------
// Group-quantized linear (4-bit weights boxed in int32, group size 128):
//   out[b,s,o] = sum_k x[b,s,k] * scale[o,g]*(q[o,k]-zero[o,g]) + bias[o]
// Mapped to a tiled f16 WMMA GEMM: M=8192, N=11008, K=4096.
// ---------------------------------------------------------------------------

typedef __attribute__((ext_vector_type(4)))  _Float16 v4h;
typedef __attribute__((ext_vector_type(8)))  _Float16 v8h;
typedef __attribute__((ext_vector_type(16))) _Float16 v16h;
typedef __attribute__((ext_vector_type(4)))  float    v4f;
typedef __attribute__((ext_vector_type(8)))  float    v8f;
typedef __attribute__((ext_vector_type(4)))  int      v4i;

namespace {
constexpr int BM  = 128;   // block tile M
constexpr int BN  = 128;   // block tile N (output features)
constexpr int BK  = 64;    // K step per LDS stage (half of quant group 128)
constexpr int LDK = 72;    // padded LDS row stride (f16 units): 144B, keeps
                           // 16B alignment for b128 and spreads banks
constexpr int GROUPS = 32; // quant groups along K (4096/128)
}

__global__ __launch_bounds__(256, 1)
void qlinear_wmma_f16(const float* __restrict__ x,
                      const int*   __restrict__ qw,
                      const float* __restrict__ scale,
                      const float* __restrict__ zero,
                      const float* __restrict__ bias,
                      float*       __restrict__ out,
                      int M, int N, int K)
{
    __shared__ _Float16 Alds[BM * LDK];   // activations, [m][k] f16
    __shared__ _Float16 Blds[BN * LDK];   // dequantized weights, [n][k] f16

    const int tid   = threadIdx.x;        // 0..255 (8 waves of 32)
    const int lane  = tid & 31;
    const int wave  = tid >> 5;           // 0..7
    const int waveM = wave & 3;           // 4 waves along M
    const int waveN = wave >> 2;          // 2 waves along N

    const int m0 = blockIdx.y * BM;
    const int n0 = blockIdx.x * BN;

    // staging coordinates: 16 threads cover one row's 64 K-values (16 x float4)
    const int tr = tid >> 4;              // 0..15 row-in-pass
    const int tc = tid & 15;              // 0..15 column (float4 granule)

    const int laneHi = lane >> 4;         // 0 or 1 (half-wave select)
    const int laneLo = lane & 15;

    // 2 (M) x 4 (N) accumulator fragments of 16x16 f32 per wave
    v8f acc[2][4];
#pragma unroll
    for (int mt = 0; mt < 2; ++mt)
#pragma unroll
        for (int nt = 0; nt < 4; ++nt)
            acc[mt][nt] = (v8f){0.f, 0.f, 0.f, 0.f, 0.f, 0.f, 0.f, 0.f};

    const float* xBase = x  + (size_t)m0 * K;
    const int*   qBase = qw + (size_t)n0 * K;

    for (int k0 = 0; k0 < K; k0 += BK) {
        // ---------------- stage A: x tile f32 -> f16 LDS ----------------
        const float* xTile = xBase + k0;
#pragma unroll
        for (int p = 0; p < 8; ++p) {
            const int row = p * 16 + tr;                    // 0..127
            v4f a4 = *(const v4f*)(xTile + (size_t)row * K + tc * 4);
            v4h h4 = { (_Float16)a4[0], (_Float16)a4[1],
                       (_Float16)a4[2], (_Float16)a4[3] };
            *(v4h*)&Alds[row * LDK + tc * 4] = h4;
        }

        // ------- stage B: qweight tile int4-in-int32 -> dequant f16 LDS -------
        const int* qTile = qBase + k0;
        const int  g     = k0 >> 7;                         // group index (gs=128)
#pragma unroll
        for (int p = 0; p < 8; ++p) {
            const int o  = p * 16 + tr;                     // 0..127 (local out idx)
            const int og = n0 + o;
            const float s = scale[og * GROUPS + g];
            const float z = zero [og * GROUPS + g];
            v4i q4 = *(const v4i*)(qTile + (size_t)o * K + tc * 4);
            v4h h4 = { (_Float16)(s * ((float)q4[0] - z)),
                       (_Float16)(s * ((float)q4[1] - z)),
                       (_Float16)(s * ((float)q4[2] - z)),
                       (_Float16)(s * ((float)q4[3] - z)) };
            *(v4h*)&Blds[o * LDK + tc * 4] = h4;
        }

        // prefetch next K-tile (global_prefetch_b8) to overlap with compute
        if (k0 + BK < K) {
            __builtin_prefetch(xTile + (size_t)tr * K + BK, 0, 1);
            __builtin_prefetch(qTile + (size_t)tr * K + BK, 0, 1);
        }

        __syncthreads();

        // ---------------- compute: 2 K-steps of 32, 8 WMMAs each ----------------
#pragma unroll
        for (int kk = 0; kk < BK; kk += 32) {
            // A fragments (ISA 16-bit A 16x32 layout):
            //   lanes 0-15: K 0..7 in V0-3, 16..23 in V4-7 (row = lane)
            //   lanes 16-31: K 8..15 and 24..31           (row = lane-16)
            v16h af[2];
#pragma unroll
            for (int mt = 0; mt < 2; ++mt) {
                const int row = waveM * 32 + mt * 16 + laneLo;
                const int ko  = kk + (laneHi << 3);
                v8h lo = *(const v8h*)&Alds[row * LDK + ko];
                v8h hi = *(const v8h*)&Alds[row * LDK + ko + 16];
                af[mt] = __builtin_shufflevector(lo, hi,
                         0,1,2,3,4,5,6,7,8,9,10,11,12,13,14,15);
            }
            // B fragments (16-bit B 32x16): lane n = laneLo; same K order as A:
            // lanes 0-15 K 0..7,16..23; lanes 16-31 K 8..15,24..31.
            v16h bf[4];
#pragma unroll
            for (int nt = 0; nt < 4; ++nt) {
                const int col = waveN * 64 + nt * 16 + laneLo;
                const int ko  = kk + (laneHi << 3);
                v8h lo = *(const v8h*)&Blds[col * LDK + ko];
                v8h hi = *(const v8h*)&Blds[col * LDK + ko + 16];
                bf[nt] = __builtin_shufflevector(lo, hi,
                         0,1,2,3,4,5,6,7,8,9,10,11,12,13,14,15);
            }
#pragma unroll
            for (int mt = 0; mt < 2; ++mt)
#pragma unroll
                for (int nt = 0; nt < 4; ++nt)
                    acc[mt][nt] = __builtin_amdgcn_wmma_f32_16x16x32_f16(
                        /*neg_a=*/false, af[mt],
                        /*neg_b=*/false, bf[nt],
                        /*c_mod=*/(short)0, acc[mt][nt],
                        /*reuse_a=*/false, /*reuse_b=*/false);
        }

        __syncthreads();
    }

    // ---------------- epilogue: bias add + store ----------------
    // C/D layout: VGPR r, lanes 0-15 -> M=r, lanes 16-31 -> M=8+r; N=lane&15.
    const int mWave = m0 + waveM * 32 + (laneHi << 3);
    const int nWave = n0 + waveN * 64 + laneLo;
#pragma unroll
    for (int mt = 0; mt < 2; ++mt) {
#pragma unroll
        for (int nt = 0; nt < 4; ++nt) {
            const int nCol = nWave + nt * 16;
            const float bv = bias[nCol];
            const int mRow = mWave + mt * 16;
            v8f c = acc[mt][nt];
#pragma unroll
            for (int r = 0; r < 8; ++r)
                out[(size_t)(mRow + r) * N + nCol] = c[r] + bv;
        }
    }
}

extern "C" void kernel_launch(void* const* d_in, const int* in_sizes, int n_in,
                              void* d_out, int out_size, void* d_ws, size_t ws_size,
                              hipStream_t stream) {
    const float* x     = (const float*)d_in[0];   // [4,2048,4096] f32
    const int*   qw    = (const int*)  d_in[1];   // [11008,4096] int32 (0..15)
    const float* scale = (const float*)d_in[2];   // [11008,32]
    const float* zero  = (const float*)d_in[3];   // [11008,32]
    const float* bias  = (const float*)d_in[4];   // [11008]
    float*       out   = (float*)d_out;           // [4,2048,11008] f32

    const int K = 4096;
    const int N = 11008;
    const int M = in_sizes[0] / K;                // 8192

    dim3 grid(N / BN, M / BM);                    // (86, 64)
    qlinear_wmma_f16<<<grid, 256, 0, stream>>>(x, qw, scale, zero, bias, out,
                                               M, N, K);
}
